// ResonanceLoss_15384572854803
// MI455X (gfx1250) — compile-verified, matrix operand order checked by reading, not run on hardware
//
#include <hip/hip_runtime.h>
#include <cmath>

typedef __attribute__((ext_vector_type(16))) _Float16 v16h;
typedef __attribute__((ext_vector_type(8)))  float    v8f;

#define TWO_PI_F 6.2831853071795864769f
#define PI_F     3.1415926535897932385f

// ---------------------------------------------------------------------------
// Zero the two workspace accumulators (ce_sum, resonance_sum) every call.
// ---------------------------------------------------------------------------
__global__ void init_ws_kernel(float* ws) {
    if (threadIdx.x < 2) ws[threadIdx.x] = 0.0f;
}

// ---------------------------------------------------------------------------
// Cross-entropy: each wave handles 16 rows of outputs[B, C].
// Row-sum of exp(x - rowmax) is computed with v_wmma_f32_16x16x32_f16:
//   A = 16x32 tile of exp values (f16), B = 32x16 ones -> every C/D element
//   of row M equals the row sum (f32 accumulate inside the WMMA).
// ---------------------------------------------------------------------------
__global__ __launch_bounds__(256) void ce_kernel(
        const float* __restrict__ outputs,
        const int*   __restrict__ targets,
        float*       __restrict__ ws,
        int B, int C) {
    __shared__ float s_sum[8][16];

    const int lane    = threadIdx.x & 31;
    const int wave    = threadIdx.x >> 5;
    const int rowbase = (blockIdx.x * 8 + wave) * 16;
    const int m       = lane & 15;                // row within the 16-row tile
    const int row     = rowbase + m;
    const bool active = (rowbase + 16 <= B);      // wave-uniform
    const float* rowp = outputs + (size_t)row * (size_t)C;

    float rmax = -INFINITY;
    v8f   acc  = {0.f, 0.f, 0.f, 0.f, 0.f, 0.f, 0.f, 0.f};

    if (active) {
        // ---- Phase 1: per-row max (2 lanes per row, float4 strided) ----
        for (int c = (lane >> 4) * 4; c + 4 <= C; c += 8) {
            float4 v = *reinterpret_cast<const float4*>(rowp + c);
            rmax = fmaxf(rmax, fmaxf(fmaxf(v.x, v.y), fmaxf(v.z, v.w)));
        }
        rmax = fmaxf(rmax, __shfl_xor(rmax, 16, 32));

        // ---- Phase 2: sum of exp via WMMA (A * ones) ----
        v16h ones;
        #pragma unroll
        for (int i = 0; i < 16; ++i) ones[i] = (_Float16)1.0f;

        // 16-bit A-matrix layout (16x32): lanes 0-15 hold K = 0..7 and 16..23,
        // lanes 16-31 hold K = 8..15 and 24..31, for row M = lane & 15.
        const int c0 = (lane >> 4) * 8;

        int base = 0;
        for (; base + 32 <= C; base += 32) {
            float4 g0a = *reinterpret_cast<const float4*>(rowp + base + c0);
            float4 g0b = *reinterpret_cast<const float4*>(rowp + base + c0 + 4);
            float4 g1a = *reinterpret_cast<const float4*>(rowp + base + 16 + c0);
            float4 g1b = *reinterpret_cast<const float4*>(rowp + base + 16 + c0 + 4);
            __builtin_prefetch(rowp + base + 32 + c0, 0, 1);   // global_prefetch_b8

            v16h a;
            a[0]  = (_Float16)__expf(g0a.x - rmax);
            a[1]  = (_Float16)__expf(g0a.y - rmax);
            a[2]  = (_Float16)__expf(g0a.z - rmax);
            a[3]  = (_Float16)__expf(g0a.w - rmax);
            a[4]  = (_Float16)__expf(g0b.x - rmax);
            a[5]  = (_Float16)__expf(g0b.y - rmax);
            a[6]  = (_Float16)__expf(g0b.z - rmax);
            a[7]  = (_Float16)__expf(g0b.w - rmax);
            a[8]  = (_Float16)__expf(g1a.x - rmax);
            a[9]  = (_Float16)__expf(g1a.y - rmax);
            a[10] = (_Float16)__expf(g1a.z - rmax);
            a[11] = (_Float16)__expf(g1a.w - rmax);
            a[12] = (_Float16)__expf(g1b.x - rmax);
            a[13] = (_Float16)__expf(g1b.y - rmax);
            a[14] = (_Float16)__expf(g1b.z - rmax);
            a[15] = (_Float16)__expf(g1b.w - rmax);

            acc = __builtin_amdgcn_wmma_f32_16x16x32_f16(
                false, a, false, ones, (short)0, acc, false, false);
        }

        // ---- Tail chunk: branch-free clamped loads + select (-inf -> 0) ----
        if (base < C) {
            v16h a;
            #pragma unroll
            for (int i = 0; i < 8; ++i) {
                int c1  = base + c0 + i;
                int cc1 = (c1 < C) ? c1 : (C - 1);       // always in-bounds
                float x1 = rowp[cc1];
                x1 = (c1 < C) ? x1 : -INFINITY;          // v_cndmask, no branch
                a[i] = (_Float16)__expf(x1 - rmax);

                int c2  = base + 16 + c0 + i;
                int cc2 = (c2 < C) ? c2 : (C - 1);
                float x2 = rowp[cc2];
                x2 = (c2 < C) ? x2 : -INFINITY;
                a[8 + i] = (_Float16)__expf(x2 - rmax);
            }
            acc = __builtin_amdgcn_wmma_f32_16x16x32_f16(
                false, a, false, ones, (short)0, acc, false, false);
        }

        // C/D layout: VGPR v, lanes 0-15 -> M=v ; lanes 16-31 -> M=v+8.
        // Every N column holds the identical row sum; stage through LDS.
        if (lane == 0) {
            #pragma unroll
            for (int i = 0; i < 8; ++i) s_sum[wave][i] = acc[i];
        }
        if (lane == 16) {
            #pragma unroll
            for (int i = 0; i < 8; ++i) s_sum[wave][8 + i] = acc[i];
        }
    }

    __syncthreads();

    float ce = 0.0f;
    if (active && lane < 16) {
        float sumexp = s_sum[wave][lane];
        int   t      = targets[row];
        float xt     = rowp[t];
        // -logp[target] = rowmax + log(sum exp(x - rowmax)) - x[target]
        ce = rmax + __logf(sumexp) - xt;
    }
    // Reduce lanes 0..15 (lanes 16..31 carry 0)
    for (int off = 8; off; off >>= 1) ce += __shfl_xor(ce, off, 32);
    if (lane == 0 && active) atomicAdd(ws, ce);
}

// ---------------------------------------------------------------------------
// Resonance: gather phase[src,dst] and phase[dst,src], wrap |diff| to [0,pi].
// ---------------------------------------------------------------------------
__global__ __launch_bounds__(256) void resonance_kernel(
        const float* __restrict__ phase,
        const int*   __restrict__ esrc,
        const int*   __restrict__ edst,
        float*       __restrict__ ws,
        int E, int N) {
    int tid    = blockIdx.x * blockDim.x + threadIdx.x;
    int stride = gridDim.x * blockDim.x;
    float sum = 0.0f;
    for (int e = tid; e < E; e += stride) {
        int s = esrc[e];
        int d = edst[e];
        float pij = phase[(size_t)s * (size_t)N + (size_t)d];
        float pji = phase[(size_t)d * (size_t)N + (size_t)s];
        float w = fmodf(fabsf(pij - pji), TWO_PI_F);
        w = (w > PI_F) ? (TWO_PI_F - w) : w;
        sum += w;
    }
    for (int off = 16; off; off >>= 1) sum += __shfl_xor(sum, off, 32);
    if ((threadIdx.x & 31) == 0) atomicAdd(ws + 1, sum);
}

// ---------------------------------------------------------------------------
// Combine: out = ce_sum/B + 0.1 * res_sum/E
// ---------------------------------------------------------------------------
__global__ void finalize_kernel(const float* __restrict__ ws,
                                float* __restrict__ out,
                                float invB, float invE) {
    if (blockIdx.x == 0 && threadIdx.x == 0) {
        out[0] = ws[0] * invB + 0.1f * (ws[1] * invE);
    }
}

// ---------------------------------------------------------------------------
extern "C" void kernel_launch(void* const* d_in, const int* in_sizes, int n_in,
                              void* d_out, int out_size, void* d_ws, size_t ws_size,
                              hipStream_t stream) {
    const float* outputs = (const float*)d_in[0];
    const int*   targets = (const int*)  d_in[1];
    const float* phase   = (const float*)d_in[2];
    const int*   esrc    = (const int*)  d_in[3];
    const int*   edst    = (const int*)  d_in[4];

    const int B = in_sizes[1];                       // 16384
    const int C = in_sizes[0] / B;                   // 1000
    const int E = in_sizes[3];                       // 131072
    const int N = (int)llround(sqrt((double)in_sizes[2]));  // 4096

    float* ws  = (float*)d_ws;
    float* out = (float*)d_out;

    init_ws_kernel<<<1, 64, 0, stream>>>(ws);

    // 8 waves/block * 16 rows/wave = 128 rows per block
    int ce_blocks = (B + 127) / 128;
    ce_kernel<<<ce_blocks, 256, 0, stream>>>(outputs, targets, ws, B, C);

    resonance_kernel<<<256, 256, 0, stream>>>(phase, esrc, edst, ws, E, N);

    finalize_kernel<<<1, 64, 0, stream>>>(ws, out, 1.0f / (float)B, 1.0f / (float)E);
}
